// MLA_36292473651444
// MI455X (gfx1250) — matrix operand order repaired and checked by reading, not compiled, not run on hardware
//
#include <hip/hip_runtime.h>

// ---------------------------------------------------------------------------
// MLA (multi-head latent attention) for MI455X / gfx1250, wave32 + WMMA bf16.
// bf16 convert -> async-LDS double-buffered WMMA GEMMs -> flash attention
// (async K staging, online softmax) -> output GEMM (fp32).
// ---------------------------------------------------------------------------

typedef __bf16 bf16_t;
typedef __attribute__((ext_vector_type(8)))  __bf16 bf16x8;
typedef __attribute__((ext_vector_type(16))) __bf16 bf16x16;
typedef __attribute__((ext_vector_type(8)))  float  f32x8;
typedef int v4i_b128 __attribute__((vector_size(16)));  // matches async builtin pointee

static constexpr int Bc = 2, Sc = 2048, Dc = 2048, Hc = 16, DKc = 128, DVc = 128, Lc = 64;

#define DEV_ static __device__ __forceinline__

// ---- CDNA5 async global->LDS path (guarded; sync fallback keeps build safe)
#if defined(__has_builtin)
#if __has_builtin(__builtin_amdgcn_global_load_async_to_lds_b128)
#define USE_ASYNC_LDS 1
#endif
#endif
#ifndef USE_ASYNC_LDS
#define USE_ASYNC_LDS 0
#endif

DEV_ void copy16_g2l(const bf16_t* g, bf16_t* l) {
#if USE_ASYNC_LDS
  __builtin_amdgcn_global_load_async_to_lds_b128(
      (__attribute__((address_space(1))) v4i_b128*)(bf16_t*)g,
      (__attribute__((address_space(3))) v4i_b128*)l, 0, 0);
#else
  *(bf16x8*)l = *(const bf16x8*)g;
#endif
}

DEV_ void wait_async0() {
#if USE_ASYNC_LDS
#if __has_builtin(__builtin_amdgcn_s_wait_asynccnt)
  __builtin_amdgcn_s_wait_asynccnt(0);
#else
  asm volatile("s_wait_asynccnt 0x0" ::: "memory");
#endif
#endif
}

DEV_ bf16x16 load_frag16(const bf16_t* p0, const bf16_t* p1) {
  bf16x8 lo = *(const bf16x8*)p0;
  bf16x8 hi = *(const bf16x8*)p1;
  return __builtin_shufflevector(lo, hi, 0,1,2,3,4,5,6,7,8,9,10,11,12,13,14,15);
}

DEV_ f32x8 wmma_bf16(bf16x16 a, bf16x16 b, f32x8 c) {
  // D = A(16x32) * B(32x16) + C, fp32 accumulate
  return __builtin_amdgcn_wmma_f32_16x16x32_bf16(false, a, false, b, (short)0, c, false, false);
}

// ------------------------------- converters --------------------------------

__global__ void cvt_f32_to_bf16(const float* __restrict__ src, bf16_t* __restrict__ dst, long n) {
  long i = (long)blockIdx.x * blockDim.x + threadIdx.x;
  long stride = (long)gridDim.x * blockDim.x;
  for (; i < n; i += stride) dst[i] = (bf16_t)__builtin_nontemporal_load(&src[i]);
}

// src[R][C] (fp32) -> dst[C][R] (bf16)
__global__ void cvt_transpose_f32_to_bf16(const float* __restrict__ src, bf16_t* __restrict__ dst,
                                          int R, int C) {
  long n = (long)R * C;
  long stride = (long)gridDim.x * blockDim.x;
  for (long i = (long)blockIdx.x * blockDim.x + threadIdx.x; i < n; i += stride) {
    long r = i / C;
    long c = i - r * C;
    dst[c * R + r] = (bf16_t)__builtin_nontemporal_load(&src[i]);
  }
}

// ------------------------------ generic GEMM -------------------------------
// C[M,N] = A[M,K] * Bt[N,K]^T + bias[N]
//   A, Bt bf16 row-major; requires M%128==0, N%128==0, K%32==0.
//   split_shift > 0: rows are (b,s) with s = low s_shift bits, cols are (h,d)
//     with d = low split_shift bits; out idx = ((b*(N>>split)+h)<<s_shift|s)<<split|d
//   split_shift == 0: plain row-major output.
// 256 threads = 8 waves; tile 128x128x32, double-buffered async LDS staging.

template <bool OUT_BF16>
__global__ __launch_bounds__(256)
void gemm_bf16nt(const bf16_t* __restrict__ A, const bf16_t* __restrict__ Bt,
                 const float* __restrict__ bias, void* __restrict__ Cout,
                 int M, int N, int K, int split_shift, int s_shift) {
  constexpr int BK = 32, PAD = 8, LDT = BK + PAD;
  __shared__ __align__(16) bf16_t As[2][128 * LDT];   // 2 x 10KB
  __shared__ __align__(16) bf16_t Bs[2][128 * LDT];   // 2 x 10KB

  const int tid  = threadIdx.x;
  const int wid  = tid >> 5;
  const int lane = tid & 31;
  const int half = lane >> 4;
  const int l15  = lane & 15;
  const int wm   = wid & 3;   // 4 m-waves x 32 rows
  const int wn   = wid >> 2;  // 2 n-waves x 64 cols
  const int m0   = blockIdx.y * 128;
  const int n0   = blockIdx.x * 128;

  const f32x8 zero8 = {0, 0, 0, 0, 0, 0, 0, 0};
  f32x8 acc[2][4];
#pragma unroll
  for (int mi = 0; mi < 2; ++mi)
#pragma unroll
    for (int ni = 0; ni < 4; ++ni) acc[mi][ni] = zero8;

  const int lr = tid >> 2;       // 0..63
  const int lc = (tid & 3) * 8;  // 0,8,16,24

  auto stage = [&](int k0, int bufIdx) {
#pragma unroll
    for (int p = 0; p < 2; ++p) {
      const int row = lr + p * 64;
      copy16_g2l(&A [(size_t)(m0 + row) * K + k0 + lc], &As[bufIdx][row * LDT + lc]);
      copy16_g2l(&Bt[(size_t)(n0 + row) * K + k0 + lc], &Bs[bufIdx][row * LDT + lc]);
    }
  };

  const int nT = K / BK;
  int buf = 0;
  stage(0, 0);

  for (int t = 0; t < nT; ++t) {
    wait_async0();        // my loads for current buffer are in LDS
    __syncthreads();      // everyone's loads done; everyone done reading buf^1
    if (t + 1 < nT) stage((t + 1) * BK, buf ^ 1);

    bf16x16 af[2], bfr[4];
#pragma unroll
    for (int mi = 0; mi < 2; ++mi) {
      const bf16_t* base = &As[buf][(wm * 32 + mi * 16 + l15) * LDT];
      af[mi] = load_frag16(base + half * 16, base + half * 16 + 8);
    }
#pragma unroll
    for (int ni = 0; ni < 4; ++ni) {
      const bf16_t* base = &Bs[buf][(wn * 64 + ni * 16 + l15) * LDT + half * 16];
      bfr[ni] = load_frag16(base, base + 8);
    }
#pragma unroll
    for (int mi = 0; mi < 2; ++mi)
#pragma unroll
      for (int ni = 0; ni < 4; ++ni) acc[mi][ni] = wmma_bf16(af[mi], bfr[ni], acc[mi][ni]);
    buf ^= 1;
  }

  // epilogue: C layout VGPR r -> row = r + 8*half, col = l15
#pragma unroll
  for (int mi = 0; mi < 2; ++mi)
#pragma unroll
    for (int ni = 0; ni < 4; ++ni) {
      const int col = n0 + wn * 64 + ni * 16 + l15;
      const float bv = bias ? bias[col] : 0.0f;
#pragma unroll
      for (int r = 0; r < 8; ++r) {
        const int row = m0 + wm * 32 + mi * 16 + r + 8 * half;
        const float v = acc[mi][ni][r] + bv;
        size_t idx;
        if (split_shift > 0) {  // power-of-two head split: shifts only
          const int bb = row >> s_shift;
          const int ss = row & ((1 << s_shift) - 1);
          const int hh = col >> split_shift;
          const int dd = col & ((1 << split_shift) - 1);
          const int nh = N >> split_shift;
          idx = ((((size_t)(bb * nh + hh) << s_shift) + ss) << split_shift) + dd;
        } else {
          idx = (size_t)row * N + col;
        }
        if (OUT_BF16) ((bf16_t*)Cout)[idx] = (bf16_t)v;
        else          ((float*)Cout)[idx]  = v;
      }
    }
}

// ---------------------------- flash attention ------------------------------
// qp, kr: [B,H,S,DK] bf16;  vr: [B,H,S,DV] bf16;  o: [B,S,H*DV] bf16
// Grid: (S/128, B*H), 256 threads = 8 waves; each wave owns 16 q rows.
// Keys processed 32 at a time: K tile async-staged to LDS (shared by all
// waves), V tile transposed into LDS, causal mask + online softmax.

__global__ __launch_bounds__(256)
void mla_flash_attn(const bf16_t* __restrict__ qp, const bf16_t* __restrict__ kr,
                    const bf16_t* __restrict__ vr, bf16_t* __restrict__ o) {
  constexpr int PAD = 8, LDT = 32 + PAD, LDTK = DKc + PAD;
  __shared__ __align__(16) bf16_t Ks[32 * LDTK];       // K tile [key][dk] (8.5KB)
  __shared__ __align__(16) bf16_t Vt[DVc * LDT];       // V tile transposed [dv][key]
  __shared__ __align__(16) bf16_t Pbuf[8 * 16 * LDT];  // per-wave P [16][32]

  const int tid  = threadIdx.x;
  const int wid  = tid >> 5;
  const int lane = tid & 31;
  const int half = lane >> 4;
  const int l15  = lane & 15;
  const int bh   = blockIdx.y;
  const int b    = bh / Hc;
  const int h    = bh - b * Hc;
  const int q0   = blockIdx.x * 128;

  const bf16_t* qbase = qp + (size_t)bh * Sc * DKc;
  const bf16_t* kbase = kr + (size_t)bh * Sc * DKc;
  const bf16_t* vbase = vr + (size_t)bh * Sc * DVc;

  // this wave's Q fragments, loaded once (A layout: M = l15, K split by half)
  const int qrowA = q0 + wid * 16 + l15;
  bf16x16 qf[4];
#pragma unroll
  for (int ks = 0; ks < 4; ++ks) {
    const bf16_t* p = qbase + (size_t)qrowA * DKc + ks * 32;
    qf[ks] = load_frag16(p + half * 16, p + half * 16 + 8);
  }

  const f32x8 zero8 = {0, 0, 0, 0, 0, 0, 0, 0};
  f32x8 oacc[8];
  float m_run[8], l_run[8];
#pragma unroll
  for (int n = 0; n < 8; ++n) oacc[n] = zero8;
#pragma unroll
  for (int r = 0; r < 8; ++r) { m_run[r] = -__builtin_inff(); l_run[r] = 0.0f; }

  const float rscale = 0.08838834764831845f;  // 1/sqrt(DK)
  bf16_t* pmine = &Pbuf[wid * 16 * LDT];
  const int nkb = q0 / 32 + 4;  // causal: keys <= q0+127

  for (int kb = 0; kb < nkb; ++kb) {
    const int kstart = kb * 32;

    __syncthreads();  // everyone done reading previous Ks/Vt
    // async-stage K tile [32 keys][128 dk] (already in B^T fragment layout)
#pragma unroll
    for (int p = 0; p < 2; ++p) {
      const int c = tid + p * 256;       // 512 x 16B chunks
      const int krow = c >> 4;           // 0..31
      const int kcol = (c & 15) * 8;     // 0..120
      copy16_g2l(&kbase[(size_t)(kstart + krow) * DKc + kcol], &Ks[krow * LDTK + kcol]);
    }
    // stage V tile transposed (VGPR path: scatter b16 into LDS)
    {
      const int k   = tid >> 3;        // 0..31
      const int dvb = (tid & 7) * 16;  // 0..112
      const bf16_t* vp = vbase + (size_t)(kstart + k) * DVc + dvb;
      bf16x8 v0 = *(const bf16x8*)vp;
      bf16x8 v1 = *(const bf16x8*)(vp + 8);
#pragma unroll
      for (int i = 0; i < 8; ++i) Vt[(dvb + i) * LDT + k] = v0[i];
#pragma unroll
      for (int i = 0; i < 8; ++i) Vt[(dvb + 8 + i) * LDT + k] = v1[i];
    }
    wait_async0();
    __syncthreads();

    // scores: S(16x32) = Q(16x128) * K^T (K fragments from LDS)
    f32x8 sc[2];
    sc[0] = zero8; sc[1] = zero8;
#pragma unroll
    for (int f = 0; f < 2; ++f) {
#pragma unroll
      for (int ks = 0; ks < 4; ++ks) {
        const bf16_t* p = &Ks[(f * 16 + l15) * LDTK + ks * 32 + half * 16];
        bf16x16 kf = load_frag16(p, p + 8);
        sc[f] = wmma_bf16(qf[ks], kf, sc[f]);
      }
    }

    // scale + causal mask, per-row max over the 32 keys
    float rmax[8];
#pragma unroll
    for (int r = 0; r < 8; ++r) {
      const int qrow = q0 + wid * 16 + r + 8 * half;
#pragma unroll
      for (int f = 0; f < 2; ++f) {
        const int key = kstart + f * 16 + l15;
        const float v = sc[f][r] * rscale;
        sc[f][r] = (key > qrow) ? -__builtin_inff() : v;
      }
      rmax[r] = fmaxf(sc[0][r], sc[1][r]);
    }
#pragma unroll
    for (int r = 0; r < 8; ++r)
#pragma unroll
      for (int off = 1; off < 16; off <<= 1)
        rmax[r] = fmaxf(rmax[r], __shfl_xor(rmax[r], off, 32));

    // online softmax update
    float alpha[8], rsum[8];
#pragma unroll
    for (int r = 0; r < 8; ++r) {
      const float mn = fmaxf(m_run[r], rmax[r]);
      alpha[r] = __expf(m_run[r] - mn);
      m_run[r] = mn;
      const float p0 = __expf(sc[0][r] - mn);
      const float p1 = __expf(sc[1][r] - mn);
      sc[0][r] = p0; sc[1][r] = p1;
      rsum[r] = p0 + p1;
    }
#pragma unroll
    for (int r = 0; r < 8; ++r) {
#pragma unroll
      for (int off = 1; off < 16; off <<= 1) rsum[r] += __shfl_xor(rsum[r], off, 32);
      l_run[r] = l_run[r] * alpha[r] + rsum[r];
    }
#pragma unroll
    for (int n = 0; n < 8; ++n)
#pragma unroll
      for (int r = 0; r < 8; ++r) oacc[n][r] *= alpha[r];

    // P: C layout -> bf16 -> per-wave LDS -> reload in A layout
#pragma unroll
    for (int f = 0; f < 2; ++f)
#pragma unroll
      for (int r = 0; r < 8; ++r)
        pmine[(r + 8 * half) * LDT + f * 16 + l15] = (bf16_t)sc[f][r];
    bf16x16 pf;
    {
      const bf16_t* pb = pmine + l15 * LDT;
      pf = load_frag16(pb + half * 16, pb + half * 16 + 8);
    }

    // O += P(16x32) * V(32x128)
#pragma unroll
    for (int n = 0; n < 8; ++n) {
      const bf16_t* vp2 = &Vt[(n * 16 + l15) * LDT + half * 16];
      bf16x16 vf = load_frag16(vp2, vp2 + 8);
      oacc[n] = wmma_bf16(pf, vf, oacc[n]);
    }
  }

  // normalize and emit as [B,S,H*DV] bf16 for the output projection
#pragma unroll
  for (int n = 0; n < 8; ++n)
#pragma unroll
    for (int r = 0; r < 8; ++r) {
      const int s = q0 + wid * 16 + r + 8 * half;
      const float inv = 1.0f / l_run[r];
      const size_t idx = (((size_t)b * Sc + s) * Hc + h) * (size_t)DVc + n * 16 + l15;
      o[idx] = (bf16_t)(oacc[n][r] * inv);
    }
}

// ------------------------------ orchestration ------------------------------

extern "C" void kernel_launch(void* const* d_in, const int* in_sizes, int n_in,
                              void* d_out, int out_size, void* d_ws, size_t ws_size,
                              hipStream_t stream) {
  (void)in_sizes; (void)n_in; (void)out_size; (void)ws_size;

  const float* queries = (const float*)d_in[0];
  const float* keys    = (const float*)d_in[1];
  const float* values  = (const float*)d_in[2];
  const float* Wq      = (const float*)d_in[3];
  const float* bq      = (const float*)d_in[4];
  const float* Wlk     = (const float*)d_in[5];
  const float* blk     = (const float*)d_in[6];
  const float* Wlv     = (const float*)d_in[7];
  const float* blv     = (const float*)d_in[8];
  const float* Wkr     = (const float*)d_in[9];
  const float* Wvr     = (const float*)d_in[10];
  const float* Wout    = (const float*)d_in[11];
  const float* bout    = (const float*)d_in[12];
  float* out = (float*)d_out;

  // bump allocator over workspace (256B aligned regions)
  char* ws = (char*)d_ws;
  size_t off = 0;
  auto alloc = [&](size_t elems) {
    bf16_t* p = (bf16_t*)(ws + off);
    off += ((elems * sizeof(bf16_t) + 255) / 256) * 256;
    return p;
  };

  const size_t BS = (size_t)Bc * Sc;             // 4096 rows
  bf16_t* qb    = alloc(BS * Dc);                // queries bf16
  bf16_t* kbuf  = alloc(BS * Dc);                // keys bf16
  bf16_t* vbuf  = alloc(BS * Dc);                // values bf16
  bf16_t* WqT   = alloc((size_t)Hc * DKc * Dc);  // [2048][2048]
  bf16_t* WlkT  = alloc((size_t)Hc * Lc * Dc);   // [1024][2048]
  bf16_t* WlvT  = alloc((size_t)Hc * Lc * Dc);
  bf16_t* WkrT  = alloc((size_t)DKc * Lc);       // [128][64]
  bf16_t* WvrT  = alloc((size_t)DVc * Lc);
  bf16_t* WoutT = alloc((size_t)Dc * Hc * DVc);  // [2048][2048]
  bf16_t* qproj = alloc(BS * Hc * DKc);          // [B,H,S,DK]
  bf16_t* lkb   = alloc(BS * Hc * Lc);           // [B,H,S,L]
  bf16_t* lvb   = alloc(BS * Hc * Lc);
  bf16_t* krec  = alloc(BS * Hc * DKc);          // [B,H,S,DK]
  bf16_t* vrec  = alloc(BS * Hc * DVc);          // [B,H,S,DV]
  bf16_t* oflat = alloc(BS * Hc * DVc);          // [B,S,H*DV]

  const dim3 t256(256);
  const long nqkv = (long)BS * Dc;

  // 1) precision conversion
  cvt_f32_to_bf16<<<dim3(2048), t256, 0, stream>>>(queries, qb,   nqkv);
  cvt_f32_to_bf16<<<dim3(2048), t256, 0, stream>>>(keys,    kbuf, nqkv);
  cvt_f32_to_bf16<<<dim3(2048), t256, 0, stream>>>(values,  vbuf, nqkv);
  cvt_transpose_f32_to_bf16<<<dim3(2048), t256, 0, stream>>>(Wq,   WqT,   Dc, Hc * DKc);
  cvt_transpose_f32_to_bf16<<<dim3(2048), t256, 0, stream>>>(Wlk,  WlkT,  Dc, Hc * Lc);
  cvt_transpose_f32_to_bf16<<<dim3(2048), t256, 0, stream>>>(Wlv,  WlvT,  Dc, Hc * Lc);
  cvt_transpose_f32_to_bf16<<<dim3(32),   t256, 0, stream>>>(Wkr,  WkrT,  Lc, DKc);
  cvt_transpose_f32_to_bf16<<<dim3(32),   t256, 0, stream>>>(Wvr,  WvrT,  Lc, DVc);
  cvt_transpose_f32_to_bf16<<<dim3(2048), t256, 0, stream>>>(Wout, WoutT, Hc * DVc, Dc);

  // 2) projections (head-split outputs; dd and S are powers of two -> shifts)
  gemm_bf16nt<true><<<dim3((Hc * DKc) / 128, (int)BS / 128), t256, 0, stream>>>(
      qb, WqT, bq, qproj, (int)BS, Hc * DKc, Dc, /*split*/7, /*s*/11);
  gemm_bf16nt<true><<<dim3((Hc * Lc) / 128, (int)BS / 128), t256, 0, stream>>>(
      kbuf, WlkT, blk, lkb, (int)BS, Hc * Lc, Dc, /*split*/6, /*s*/11);
  gemm_bf16nt<true><<<dim3((Hc * Lc) / 128, (int)BS / 128), t256, 0, stream>>>(
      vbuf, WlvT, blv, lvb, (int)BS, Hc * Lc, Dc, /*split*/6, /*s*/11);

  // 3) latent -> key/value reconstruction ([B*H*S, L] x [L, DK/DV])
  gemm_bf16nt<true><<<dim3(DKc / 128, (int)(BS * Hc) / 128), t256, 0, stream>>>(
      lkb, WkrT, nullptr, krec, (int)(BS * Hc), DKc, Lc, 0, 0);
  gemm_bf16nt<true><<<dim3(DVc / 128, (int)(BS * Hc) / 128), t256, 0, stream>>>(
      lvb, WvrT, nullptr, vrec, (int)(BS * Hc), DVc, Lc, 0, 0);

  // 4) causal flash attention
  mla_flash_attn<<<dim3(Sc / 128, Bc * Hc), t256, 0, stream>>>(qproj, krec, vrec, oflat);

  // 5) output projection (fp32 result)
  gemm_bf16nt<false><<<dim3(Dc / 128, (int)BS / 128), t256, 0, stream>>>(
      oflat, WoutT, bout, out, (int)BS, Dc, Hc * DVc, 0, 0);
}
